// Model_37271726194900
// MI455X (gfx1250) — matrix operand order repaired and checked by reading, not compiled, not run on hardware
//
#include <hip/hip_runtime.h>
#include <hip/hip_bf16.h>

#define N_NODES 50000
#define N_EDGES 400000
#define N_GRAPHS 64
#define HID 200
#define NUM_OUT 16
#define NLAYERS 5
#define MAX_EPOS 20
#define BN_EPS 1e-5f

#define KPAD 224                 // K padded to 7*32 for guard-free WMMA K-loop
#define KSTEPS 7                 // 224 / 32
#define MT (N_NODES / 16)        // 3125 (exact)
#define NT 13                    // ceil(200/16) -> cols 0..207, 200..207 zeroed
#define NTILES (MT * NT)

typedef __attribute__((ext_vector_type(16))) __bf16 v16bf;
typedef __attribute__((ext_vector_type(8)))  float  v8f;

static __device__ __forceinline__ unsigned short f2bf(float f) {
  __bf16 b = (__bf16)f;
  return __builtin_bit_cast(unsigned short, b);
}
static __device__ __forceinline__ unsigned int pk2bf(float lo, float hi) {
  return (unsigned int)f2bf(lo) | ((unsigned int)f2bf(hi) << 16);
}

// ---------------------------------------------------------------------------
// Node embedding: x = kind_emb[kind] + (kind==0 ? inst2vec[content] : type_emb)
// ---------------------------------------------------------------------------
__global__ __launch_bounds__(256) void node_embed_kernel(
    const int* __restrict__ kind, const int* __restrict__ content,
    const float* __restrict__ kind_emb, const float* __restrict__ inst2vec,
    const float* __restrict__ type_emb, float* __restrict__ x) {
  int id = blockIdx.x * blockDim.x + threadIdx.x;
  if (id >= N_NODES * (HID / 4)) return;
  int n = id / (HID / 4);
  int c = (id % (HID / 4)) * 4;
  int k = kind[n];
  const float4 kv = *(const float4*)(kind_emb + (size_t)k * HID + c);
  float4 cv;
  if (k == 0) cv = *(const float4*)(inst2vec + (size_t)content[n] * HID + c);
  else        cv = *(const float4*)(type_emb + c);
  float4 o;
  o.x = kv.x + cv.x; o.y = kv.y + cv.y; o.z = kv.z + cv.z; o.w = kv.w + cv.w;
  *(float4*)(x + (size_t)n * HID + c) = o;
}

// ---------------------------------------------------------------------------
// Edge message + scatter: agg[dst] += relu(x[src] + etype_emb + epos_emb)
// Working set (x, agg) is L2-resident (192MB); hardware f32 atomics.
// ---------------------------------------------------------------------------
__global__ __launch_bounds__(256) void msg_scatter_kernel(
    const float* __restrict__ x, const int* __restrict__ ei,
    const int* __restrict__ etype, const int* __restrict__ epos,
    const float* __restrict__ etype_emb, const float* __restrict__ epos_emb,
    float* __restrict__ agg) {
  int id = blockIdx.x * blockDim.x + threadIdx.x;
  if (id >= N_EDGES * (HID / 4)) return;
  int e = id / (HID / 4);
  int c = (id % (HID / 4)) * 4;
  int src = ei[e];
  int dst = ei[N_EDGES + e];
  int t = etype[e];
  int p = epos[e]; p = p < (MAX_EPOS - 1) ? p : (MAX_EPOS - 1);
  const float4 xv = *(const float4*)(x + (size_t)src * HID + c);
  const float4 tv = *(const float4*)(etype_emb + (size_t)t * HID + c);
  const float4 pv = *(const float4*)(epos_emb + (size_t)p * HID + c);
  float4 m;
  m.x = fmaxf(xv.x + tv.x + pv.x, 0.f);
  m.y = fmaxf(xv.y + tv.y + pv.y, 0.f);
  m.z = fmaxf(xv.z + tv.z + pv.z, 0.f);
  m.w = fmaxf(xv.w + tv.w + pv.w, 0.f);
  float* a = agg + (size_t)dst * HID + c;
  unsafeAtomicAdd(a + 0, m.x);
  unsafeAtomicAdd(a + 1, m.y);
  unsafeAtomicAdd(a + 2, m.z);
  unsafeAtomicAdd(a + 3, m.w);
}

// ---------------------------------------------------------------------------
// Pack A = (x + agg) into zero-padded bf16 [N_NODES x KPAD].
// One thread per output dword (2 bf16). K tail (200..223) written as zero, so
// the WMMA K-loop needs no bounds checks.
// ---------------------------------------------------------------------------
__global__ __launch_bounds__(256) void pack_a_kernel(
    const float* __restrict__ x, const float* __restrict__ agg,
    unsigned int* __restrict__ abf) {
  int id = blockIdx.x * blockDim.x + threadIdx.x;
  if (id >= N_NODES * (KPAD / 2)) return;
  int n = id / (KPAD / 2);
  int c = (id % (KPAD / 2)) * 2;          // even K index; 200 is even
  unsigned int v = 0u;
  if (c < HID) {
    size_t o = (size_t)n * HID + c;
    v = pk2bf(x[o] + agg[o], x[o + 1] + agg[o + 1]);
  }
  abf[id] = v;
}

// ---------------------------------------------------------------------------
// Pack W [200x200] f32 into fragment-native bf16 layout:
//   Bbf[((nt*KSTEPS + ks)*32 + lane)*16 + j]  = W[k, col]  (0 outside)
// with j=2v+odd, k = ks*32 + (j<8 ? (lane>>4)*8 + j : 16 + (lane>>4)*8 + j-8), col = nt*16 + (lane&15),
// matching the CDNA5 16-bit B 32x16 lane layout. GEMM then loads B as two
// unconditional b128s per lane. Total packed size: 93KB (L2/L0 resident).
// ---------------------------------------------------------------------------
__global__ __launch_bounds__(256) void pack_b_kernel(
    const float* __restrict__ W, unsigned int* __restrict__ bbf) {
  int id = blockIdx.x * blockDim.x + threadIdx.x;
  if (id >= NT * KSTEPS * 32 * 8) return;
  int d    = id & 7;                       // dword within fragment (pair 2d,2d+1)
  int lane = (id >> 3) & 31;
  int ks   = (id >> 8) % KSTEPS;
  int nt   = (id >> 8) / KSTEPS;
  int col  = nt * 16 + (lane & 15);
  int k0   = ks * 32 + (d >> 2) * 16 + (lane >> 4) * 8 + (d & 3) * 2;
  float w0 = 0.f, w1 = 0.f;
  if (col < HID) {
    if (k0 < HID)     w0 = W[(size_t)k0 * HID + col];
    if (k0 + 1 < HID) w1 = W[(size_t)(k0 + 1) * HID + col];
  }
  bbf[id] = pk2bf(w0, w1);
}

// ---------------------------------------------------------------------------
// BF16 WMMA GEMM over packed operands. One 16x16 tile per wave, 8 waves/blk.
// Inner loop: 2x global_load_b128 (A) + 2x global_load_b128 (B) + 1 WMMA,
// fully unrolled (7 steps), zero control flow. EXEC all-ones at the WMMA
// (early exit is wave-uniform).
//   GEMM1 (do_bn=1): epilogue bias+BN+ReLU, writes padded bf16 (feeds GEMM2).
//   GEMM2 (do_bn=0): epilogue bias+ReLU, writes f32 x (feeds scatter/pool).
// ---------------------------------------------------------------------------
__global__ __launch_bounds__(256) void gemm_wmma_kernel(
    const unsigned int* __restrict__ abf,   // [N_NODES x KPAD] bf16 as dwords
    const unsigned int* __restrict__ bbf,   // packed fragments
    const float* __restrict__ bias,
    const float* __restrict__ bng, const float* __restrict__ bnb,
    const float* __restrict__ bnm, const float* __restrict__ bnv,
    unsigned short* __restrict__ out_bf,    // GEMM1: [N_NODES x KPAD] bf16
    float* __restrict__ out_f32,            // GEMM2: [N_NODES x HID] f32
    int do_bn) {
  int wave = blockIdx.x * (blockDim.x >> 5) + (threadIdx.x >> 5);
  if (wave >= NTILES) return;               // uniform per wave
  int mt = wave / NT;
  int nt = wave - mt * NT;
  int lane = threadIdx.x & 31;
  int half = lane >> 4;
  int lm   = lane & 15;
  int colg = nt * 16 + lm;
  bool colok = colg < HID;

  // A: row = mt*16 + lm, dword offset in row: (k0 + half*8)/2 and +8
  const uint4* arow =
      (const uint4*)(abf + (size_t)(mt * 16 + lm) * (KPAD / 2) + half * 4);
  // B: lane-contiguous fragments
  const uint4* brow = (const uint4*)(bbf + ((size_t)(nt * KSTEPS) * 32 + lane) * 8);

  v8f acc = {};
  #pragma unroll
  for (int ks = 0; ks < KSTEPS; ++ks) {
    union { v16bf v; uint4 q[2]; } fa, fb;
    fa.q[0] = arow[ks * 4];                 // K = k0 + half*8 .. +7
    fa.q[1] = arow[ks * 4 + 2];             // K = k0 + 16 + half*8 .. +7
    fb.q[0] = brow[ks * 64];                // fragment dwords 0..3
    fb.q[1] = brow[ks * 64 + 1];            // fragment dwords 4..7
    acc = __builtin_amdgcn_wmma_f32_16x16x32_bf16(
        false, fa.v, false, fb.v, (short)0, acc, false, false);
  }

  float bcol = 0.f, sc = 1.f, mu = 0.f, bb = 0.f;
  if (colok) {
    bcol = bias[colg];
    if (do_bn) {
      sc = bng[colg] * rsqrtf(bnv[colg] + BN_EPS);
      mu = bnm[colg];
      bb = bnb[colg];
    }
  }
  #pragma unroll
  for (int r = 0; r < 8; ++r) {
    float v = acc[r] + bcol;
    if (do_bn) v = (v - mu) * sc + bb;
    v = fmaxf(v, 0.f);
    int orow = mt * 16 + half * 8 + r;      // C/D: lanes 16-31 hold M=r+8
    if (do_bn) {
      // bf16 padded output; cols >= 200 must be zero (they feed GEMM2's K)
      out_bf[(size_t)orow * KPAD + colg] = colok ? f2bf(v) : (unsigned short)0;
    } else if (colok) {
      out_f32[(size_t)orow * HID + colg] = v;
    }
  }
}

// ---------------------------------------------------------------------------
// Per-graph mean-pool pieces + jumping-knowledge head.
// ---------------------------------------------------------------------------
__global__ __launch_bounds__(256) void pool_kernel(
    const float* __restrict__ x, const int* __restrict__ batch,
    float* __restrict__ pooled) {
  int id = blockIdx.x * blockDim.x + threadIdx.x;
  if (id >= N_NODES * (HID / 4)) return;
  int n = id / (HID / 4);
  int c = (id % (HID / 4)) * 4;
  int g = batch[n];
  const float4 v = *(const float4*)(x + (size_t)n * HID + c);
  float* p = pooled + (size_t)g * HID + c;
  unsafeAtomicAdd(p + 0, v.x);
  unsafeAtomicAdd(p + 1, v.y);
  unsafeAtomicAdd(p + 2, v.z);
  unsafeAtomicAdd(p + 3, v.w);
}

__global__ __launch_bounds__(256) void count_kernel(
    const int* __restrict__ batch, float* __restrict__ counts) {
  int n = blockIdx.x * blockDim.x + threadIdx.x;
  if (n >= N_NODES) return;
  unsafeAtomicAdd(&counts[batch[n]], 1.0f);
}

__global__ __launch_bounds__(256) void head_kernel(
    const float* __restrict__ pooled, const float* __restrict__ counts,
    const float* __restrict__ fcW, const float* __restrict__ fcb,
    float* __restrict__ out) {
  int id = blockIdx.x * blockDim.x + threadIdx.x;
  if (id >= N_GRAPHS * NUM_OUT) return;
  int g = id >> 4;
  int o = id & 15;
  float cnt = fmaxf(counts[g], 1.0f);
  float s = 0.f;
  #pragma unroll 4
  for (int k = 0; k < HID; ++k)
    s += pooled[(size_t)g * HID + k] * fcW[(size_t)k * NUM_OUT + o];
  out[id] += s / cnt + fcb[o];
}

// ---------------------------------------------------------------------------
extern "C" void kernel_launch(void* const* d_in, const int* in_sizes, int n_in,
                              void* d_out, int out_size, void* d_ws, size_t ws_size,
                              hipStream_t stream) {
  const int*   node_kind    = (const int*)d_in[0];
  const int*   node_content = (const int*)d_in[1];
  const int*   edge_index   = (const int*)d_in[2];
  const int*   edge_type    = (const int*)d_in[3];
  const int*   edge_pos     = (const int*)d_in[4];
  const int*   batch_vec    = (const int*)d_in[5];
  const float* kind_emb     = (const float*)d_in[6];
  const float* inst2vec     = (const float*)d_in[7];
  const float* type_emb     = (const float*)d_in[8];
  const float* etype_emb    = (const float*)d_in[9];
  const float* epos_emb     = (const float*)d_in[10];
  const float* W1           = (const float*)d_in[11];
  const float* b1           = (const float*)d_in[12];
  const float* bn_g         = (const float*)d_in[13];
  const float* bn_b         = (const float*)d_in[14];
  const float* bn_m         = (const float*)d_in[15];
  const float* bn_v         = (const float*)d_in[16];
  const float* W2           = (const float*)d_in[17];
  const float* b2           = (const float*)d_in[18];
  const float* fc_W         = (const float*)d_in[19];
  const float* fc_b         = (const float*)d_in[20];
  float* out = (float*)d_out;

  // Workspace layout: x | agg | Abf | Hbf | Bbf | pooled | counts (~125 MB)
  char* p = (char*)d_ws;
  float*          x      = (float*)p;          p += (size_t)N_NODES * HID * 4;
  float*          agg    = (float*)p;          p += (size_t)N_NODES * HID * 4;
  unsigned int*   abf    = (unsigned int*)p;   p += (size_t)N_NODES * KPAD * 2;
  unsigned short* hbf    = (unsigned short*)p; p += (size_t)N_NODES * KPAD * 2;
  unsigned int*   bbf    = (unsigned int*)p;   p += (size_t)NT * KSTEPS * 32 * 8 * 4;
  float*          pooled = (float*)p;          p += (size_t)N_GRAPHS * HID * 4;
  float*          counts = (float*)p;

  const int nodeBlocks  = (N_NODES * (HID / 4) + 255) / 256;
  const int edgeBlocks  = (N_EDGES * (HID / 4) + 255) / 256;
  const int packABlocks = (N_NODES * (KPAD / 2) + 255) / 256;
  const int packBBlocks = (NT * KSTEPS * 32 * 8 + 255) / 256;
  const int gemmBlocks  = (NTILES + 7) / 8;

  hipMemsetAsync(out, 0, (size_t)N_GRAPHS * NUM_OUT * sizeof(float), stream);
  hipMemsetAsync(counts, 0, (size_t)N_GRAPHS * sizeof(float), stream);
  count_kernel<<<(N_NODES + 255) / 256, 256, 0, stream>>>(batch_vec, counts);

  node_embed_kernel<<<nodeBlocks, 256, 0, stream>>>(
      node_kind, node_content, kind_emb, inst2vec, type_emb, x);

  hipMemsetAsync(pooled, 0, (size_t)N_GRAPHS * HID * sizeof(float), stream);
  pool_kernel<<<nodeBlocks, 256, 0, stream>>>(x, batch_vec, pooled);
  head_kernel<<<(N_GRAPHS * NUM_OUT + 255) / 256, 256, 0, stream>>>(
      pooled, counts, fc_W, fc_b, out);

  for (int i = 0; i < NLAYERS; ++i) {
    hipMemsetAsync(agg, 0, (size_t)N_NODES * HID * sizeof(float), stream);
    msg_scatter_kernel<<<edgeBlocks, 256, 0, stream>>>(
        x, edge_index, edge_type, edge_pos, etype_emb, epos_emb, agg);

    // GEMM1: hbf = bf16( ReLU(BN((x+agg) @ W1 + b1)) ), K-padded
    pack_a_kernel<<<packABlocks, 256, 0, stream>>>(x, agg, abf);
    pack_b_kernel<<<packBBlocks, 256, 0, stream>>>(W1 + (size_t)i * HID * HID, bbf);
    gemm_wmma_kernel<<<gemmBlocks, 256, 0, stream>>>(
        abf, bbf, b1 + (size_t)i * HID,
        bn_g + (size_t)i * HID, bn_b + (size_t)i * HID,
        bn_m + (size_t)i * HID, bn_v + (size_t)i * HID,
        hbf, nullptr, /*do_bn=*/1);

    // GEMM2: x = ReLU(hbf @ W2 + b2)  (A already packed bf16 by GEMM1)
    pack_b_kernel<<<packBBlocks, 256, 0, stream>>>(W2 + (size_t)i * HID * HID, bbf);
    gemm_wmma_kernel<<<gemmBlocks, 256, 0, stream>>>(
        (const unsigned int*)hbf, bbf, b2 + (size_t)i * HID,
        nullptr, nullptr, nullptr, nullptr,
        nullptr, x, /*do_bn=*/0);

    hipMemsetAsync(pooled, 0, (size_t)N_GRAPHS * HID * sizeof(float), stream);
    pool_kernel<<<nodeBlocks, 256, 0, stream>>>(x, batch_vec, pooled);
    head_kernel<<<(N_GRAPHS * NUM_OUT + 255) / 256, 256, 0, stream>>>(
        pooled, counts, fc_W + (size_t)(i + 1) * HID * NUM_OUT,
        fc_b + (size_t)(i + 1) * NUM_OUT, out);
  }
}
